// RealQuantizedLinear_88570815578568
// MI455X (gfx1250) — matrix operand order repaired and checked by reading, not compiled, not run on hardware
//
#include <hip/hip_runtime.h>

typedef _Float16 v16h __attribute__((ext_vector_type(16)));
typedef _Float16 v8h  __attribute__((ext_vector_type(8)));
typedef float    v8f  __attribute__((ext_vector_type(8)));

#define M_DIM 8192
#define N_DIM 16384
#define K_DIM 4096

#define BM 128
#define BN 128
#define BK 32
#define LDSK 40                  // BK + 8 f16 padding -> conflict-free ds_load_b128
#define NSTAGE (K_DIM / BK)      // 128

__global__ __launch_bounds__(256)
void qlinear_wmma_f16(const float* __restrict__ x,
                      const int*   __restrict__ w8,
                      const int*   __restrict__ bias,
                      const float* __restrict__ scale_p,
                      float* __restrict__ out) {
  __shared__ __align__(16) _Float16 As[2][BM][LDSK];
  __shared__ __align__(16) _Float16 Bs[2][BN][LDSK];

  const int tid  = threadIdx.x;
  const int lane = tid & 31;
  const int wave = tid >> 5;
  const int lr   = lane & 15;        // row/col within 16x16 tile
  const int lh   = lane >> 4;        // half-wave selector
  const int wm   = (wave >> 1) * 32; // wave M offset in tile (4 rows of waves)
  const int wn   = (wave & 1) * 64;  // wave N offset in tile (2 cols of waves)

  const int Mbase = blockIdx.x * BM; // M fastest: 64 consecutive blocks share W strip,
  const int Nbase = blockIdx.y * BN; // x (128MB) cycles and stays L2-resident

  // global staging: each thread owns a 16-element half-row of the 128x32 slab
  const int grow = tid >> 1;
  const int gcol = (tid & 1) * 16;

  const float* xg = x  + (size_t)(Mbase + grow) * K_DIM + gcol;
  const int*   wg = w8 + (size_t)(Nbase + grow) * K_DIM + gcol;

  float4 ar[4];
  int4   br[4];

  v8f acc[2][4];
  const v8f vzero = {0.f, 0.f, 0.f, 0.f, 0.f, 0.f, 0.f, 0.f};
  #pragma unroll
  for (int a = 0; a < 2; ++a)
    #pragma unroll
    for (int b = 0; b < 4; ++b)
      acc[a][b] = vzero;

  // prologue: stage 0 global loads into registers
  {
    const float4* xa = (const float4*)xg;
    const int4*   wa = (const int4*)wg;
    ar[0] = xa[0]; ar[1] = xa[1]; ar[2] = xa[2]; ar[3] = xa[3];
    br[0] = wa[0]; br[1] = wa[1]; br[2] = wa[2]; br[3] = wa[3];
  }

  for (int s = 0; s < NSTAGE; ++s) {
    const int buf = s & 1;

    // ---- convert staged registers to f16 and store into LDS[buf] ----
    {
      v8h p0, p1;
      p0[0] = (_Float16)ar[0].x; p0[1] = (_Float16)ar[0].y;
      p0[2] = (_Float16)ar[0].z; p0[3] = (_Float16)ar[0].w;
      p0[4] = (_Float16)ar[1].x; p0[5] = (_Float16)ar[1].y;
      p0[6] = (_Float16)ar[1].z; p0[7] = (_Float16)ar[1].w;
      p1[0] = (_Float16)ar[2].x; p1[1] = (_Float16)ar[2].y;
      p1[2] = (_Float16)ar[2].z; p1[3] = (_Float16)ar[2].w;
      p1[4] = (_Float16)ar[3].x; p1[5] = (_Float16)ar[3].y;
      p1[6] = (_Float16)ar[3].z; p1[7] = (_Float16)ar[3].w;
      *(v8h*)&As[buf][grow][gcol]     = p0;   // ds_store_b128
      *(v8h*)&As[buf][grow][gcol + 8] = p1;

      v8h q0, q1;
      q0[0] = (_Float16)br[0].x; q0[1] = (_Float16)br[0].y;
      q0[2] = (_Float16)br[0].z; q0[3] = (_Float16)br[0].w;
      q0[4] = (_Float16)br[1].x; q0[5] = (_Float16)br[1].y;
      q0[6] = (_Float16)br[1].z; q0[7] = (_Float16)br[1].w;
      q1[0] = (_Float16)br[2].x; q1[1] = (_Float16)br[2].y;
      q1[2] = (_Float16)br[2].z; q1[3] = (_Float16)br[2].w;
      q1[4] = (_Float16)br[3].x; q1[5] = (_Float16)br[3].y;
      q1[6] = (_Float16)br[3].z; q1[7] = (_Float16)br[3].w;
      *(v8h*)&Bs[buf][grow][gcol]     = q0;
      *(v8h*)&Bs[buf][grow][gcol + 8] = q1;
    }
    __syncthreads();

    // ---- prefetch next K-slab into registers (hides HBM latency) ----
    if (s + 1 < NSTAGE) {
      const float4* xa = (const float4*)(xg + (size_t)(s + 1) * BK);
      const int4*   wa = (const int4*)(wg + (size_t)(s + 1) * BK);
      ar[0] = xa[0]; ar[1] = xa[1]; ar[2] = xa[2]; ar[3] = xa[3];
      br[0] = wa[0]; br[1] = wa[1]; br[2] = wa[2]; br[3] = wa[3];
    }

    // ---- WMMA fragments from LDS[buf] ----
    // A 16x32 f16: lanes 0-15 row M=lr with K{0..7,16..23}; lanes 16-31 K{8..15,24..31}
    v16h af[2];
    #pragma unroll
    for (int a = 0; a < 2; ++a) {
      const _Float16* rp = &As[buf][wm + a * 16 + lr][0];
      v8h c0 = *(const v8h*)(rp + lh * 8);        // ds_load_b128
      v8h c1 = *(const v8h*)(rp + 16 + lh * 8);
      af[a] = __builtin_shufflevector(c0, c1, 0,1,2,3,4,5,6,7,8,9,10,11,12,13,14,15);
    }
    // B 32x16 f16: lane holds column N=lr, K{0..7,16..23} for lanes 0-15; K{8..15,24..31} for lanes 16-31
    v16h bfv[4];
    #pragma unroll
    for (int b = 0; b < 4; ++b) {
      const _Float16* rp = &Bs[buf][wn + b * 16 + lr][0];
      v8h c0 = *(const v8h*)(rp + lh * 8);
      v8h c1 = *(const v8h*)(rp + 16 + lh * 8);
      bfv[b] = __builtin_shufflevector(c0, c1, 0,1,2,3,4,5,6,7,8,9,10,11,12,13,14,15);
    }

    #pragma unroll
    for (int a = 0; a < 2; ++a)
      #pragma unroll
      for (int b = 0; b < 4; ++b)
        acc[a][b] = __builtin_amdgcn_wmma_f32_16x16x32_f16(
            false, af[a], false, bfv[b], (short)0, acc[a][b], false, false);
  }

  // ---- epilogue: out = scale * (acc + bias_int); NT stores keep x/W in L2 ----
  const float scale = scale_p[0];
  #pragma unroll
  for (int b = 0; b < 4; ++b) {
    const int ng   = Nbase + wn + b * 16 + lr;
    const float bf = (float)bias[ng];
    #pragma unroll
    for (int a = 0; a < 2; ++a) {
      const int mg0 = Mbase + wm + a * 16 + lh * 8;  // C layout: VGPR j -> M=j+8*lh, N=lr
      #pragma unroll
      for (int j = 0; j < 8; ++j) {
        float v = scale * (acc[a][b][j] + bf);
        __builtin_nontemporal_store(v, out + (size_t)(mg0 + j) * N_DIM + ng);
      }
    }
  }
}

extern "C" void kernel_launch(void* const* d_in, const int* in_sizes, int n_in,
                              void* d_out, int out_size, void* d_ws, size_t ws_size,
                              hipStream_t stream) {
  const float* x     = (const float*)d_in[0];
  const int*   w8    = (const int*)d_in[1];
  const int*   bias  = (const int*)d_in[2];
  const float* scale = (const float*)d_in[3];
  float* out = (float*)d_out;

  dim3 grid(M_DIM / BM, N_DIM / BN);  // (64, 128)
  qlinear_wmma_f16<<<grid, dim3(256), 0, stream>>>(x, w8, bias, scale, out);
}
